// ExtractModel_19825569038482
// MI455X (gfx1250) — matrix operand order repaired and checked by reading, not compiled, not run on hardware
//
#include <hip/hip_runtime.h>

typedef __attribute__((ext_vector_type(16))) __bf16 v16bf;
typedef __attribute__((ext_vector_type(8)))  __bf16 v8bf;
typedef __attribute__((ext_vector_type(8)))  float  v8f;

#define CTX_W 0.2f
#define PLAIN_W 0.8f

// Problem dims
#define NV   4000
#define NL   12
#define NK   1024   // known units
#define NLO  1024   // lost units
#define ND   512    // char emb dim
#define NM   (NV*NL)      // 48000 flat (v,l) rows; 48000 % 16 == 0
#define NMT  (NM/16)      // 3000 M-tiles

#define CONV_MB 4         // M-tiles per WG in conv GEMM  (3000 % 4 == 0)
#define COST_MB 2         // M-tiles per WG in costs GEMM (3000 % 2 == 0)

__device__ __forceinline__ unsigned short f32_to_bf16(float f) {
  unsigned int u = __float_as_uint(f);
  unsigned int r = u + 0x7FFFu + ((u >> 16) & 1u);  // round-to-nearest-even
  return (unsigned short)(r >> 16);
}

// A-fragment (16x32 bf16): lane l holds row m=l&15; hi=l>>4 selects K phase.
// 16 consecutive k: k in [32*kc + 16*hi, +16)
__device__ __forceinline__ v16bf load_a_frag(const unsigned short* p) {
  v8bf lo = *(const v8bf*)(const void*)p;
  v8bf hi = *(const v8bf*)(const void*)(p + 8);
  return __builtin_shufflevector(lo, hi, 0,1,2,3,4,5,6,7,8,9,10,11,12,13,14,15);
}
// B-fragment from B^T row-major [N, K]: lane l holds col n=l&15,
// 16 consecutive k at 32*kc + 16*(l>>4): one contiguous 32B load.
__device__ __forceinline__ v16bf load_b_frag(const unsigned short* p) {
  return *(const v16bf*)(const void*)p;
}

__device__ __forceinline__ v8f wmma_bf16(v16bf a, v16bf b, v8f c) {
  return __builtin_amdgcn_wmma_f32_16x16x32_bf16(false, a, false, b, (short)0, c, false, false);
}

// ---------------- prep kernels ----------------
__global__ void k_f32_to_bf16(const float* __restrict__ s, unsigned short* __restrict__ d, int n) {
  int i = blockIdx.x * blockDim.x + threadIdx.x;
  if (i < n) d[i] = f32_to_bf16(s[i]);
}

// kueT[d*NK + k] = kue[k*ND + d]  (B^T for the lue GEMM)
__global__ void k_transpose_kue(const float* __restrict__ kue, unsigned short* __restrict__ kueT) {
  int i = blockIdx.x * blockDim.x + threadIdx.x;
  if (i >= ND * NK) return;
  int d = i / NK, k = i - d * NK;
  kueT[i] = f32_to_bf16(kue[(size_t)k * ND + d]);
}

// wk[(tap*ND + n)*ND + k] = w[n*ND*3 + k*3 + tap]   (B^T per tap: [d_out][d_in])
__global__ void k_pack_w(const float* __restrict__ w, unsigned short* __restrict__ wk) {
  int i = blockIdx.x * blockDim.x + threadIdx.x;
  if (i >= 3 * ND * ND) return;
  int tap = i / (ND * ND);
  int rem = i - tap * ND * ND;
  int n = rem / ND, k = rem - n * ND;
  wk[i] = f32_to_bf16(w[((size_t)n * ND + k) * 3 + tap]);
}

// e_lin: [NV][14][ND] bf16; q=0 and q=13 zero pads, q=l+1 = masked embedding row.
__global__ void k_build_elin(const unsigned short* __restrict__ kue_bf,
                             const int* __restrict__ seg, const int* __restrict__ vlen,
                             unsigned short* __restrict__ e_lin) {
  int row = blockIdx.x;              // 0 .. NV*14-1
  int v = row / 14, q = row - v * 14;
  unsigned int* dst = (unsigned int*)(e_lin + (size_t)row * ND);
  unsigned int val = 0u;
  if (q != 0 && q != 13) {
    int l = q - 1;
    if (l < vlen[v]) {
      int s = seg[v * NL + l];
      const unsigned int* src = (const unsigned int*)(kue_bf + (size_t)s * ND);
      val = src[threadIdx.x];
    }
  }
  dst[threadIdx.x] = val;            // 256 threads x 4B = 512 bf16
}

// ---------------- GEMM: lost_unit_emb = aligner[LO,K] @ kue[K,D] -> bf16 [LO,D] ----------------
__global__ void __launch_bounds__(256) k_gemm_lue(const unsigned short* __restrict__ A,   // ua bf16 [NLO,NK]
                                                 const unsigned short* __restrict__ BT,  // kueT bf16 [ND,NK]
                                                 unsigned short* __restrict__ lue) {     // [NLO,ND]
  int wave = threadIdx.x >> 5, lane = threadIdx.x & 31;
  int tile = blockIdx.x * 8 + wave;            // 2048 tiles total
  int mt = tile >> 5, nt = tile & 31;          // 64 x 32
  int mA = lane & 15, hi = lane >> 4;
  v8f acc = {0.f,0.f,0.f,0.f,0.f,0.f,0.f,0.f};
#pragma unroll 4
  for (int kc = 0; kc < NK / 32; ++kc) {
    v16bf a = load_a_frag(A + (size_t)(mt * 16 + mA) * NK + kc * 32 + hi * 16);
    v16bf b = load_b_frag(BT + (size_t)(nt * 16 + mA) * NK + kc * 32 + hi * 16);
    acc = wmma_bf16(a, b, acc);
  }
  int n = nt * 16 + (lane & 15);
#pragma unroll
  for (int r = 0; r < 8; ++r) {
    int m = mt * 16 + r + 8 * hi;
    lue[(size_t)m * ND + n] = f32_to_bf16(acc[r]);
  }
}

// ---------------- unit_logits + log_softmax + alignment ----------------
__global__ void __launch_bounds__(256) k_unit_logprobs(const unsigned short* __restrict__ kue_bf, // [NK,ND]
                                                       const unsigned short* __restrict__ lue,    // [NLO,ND]
                                                       float* __restrict__ ulp,                   // [NK,NLO]
                                                       float* __restrict__ align_out) {           // [NK,NLO]
  __shared__ float red[8][16];
  __shared__ float rowmax[16];
  __shared__ float lse[16];
  int wave = threadIdx.x >> 5, lane = threadIdx.x & 31;
  int mA = lane & 15, hi = lane >> 4;
  v8f acc[8];
  v8f z = {0.f,0.f,0.f,0.f,0.f,0.f,0.f,0.f};
#pragma unroll
  for (int t = 0; t < 8; ++t) acc[t] = z;
  for (int kc = 0; kc < ND / 32; ++kc) {
    v16bf a = load_a_frag(kue_bf + (size_t)(blockIdx.x * 16 + mA) * ND + kc * 32 + hi * 16);
#pragma unroll
    for (int t = 0; t < 8; ++t) {
      int nt = wave * 8 + t;
      v16bf b = load_b_frag(lue + (size_t)(nt * 16 + mA) * ND + kc * 32 + hi * 16);
      acc[t] = wmma_bf16(a, b, acc[t]);
    }
  }
  // row max
  float pm[8];
#pragma unroll
  for (int r = 0; r < 8; ++r) {
    float m = acc[0][r];
#pragma unroll
    for (int t = 1; t < 8; ++t) m = fmaxf(m, acc[t][r]);
    for (int off = 1; off < 16; off <<= 1) m = fmaxf(m, __shfl_xor(m, off, 32));
    pm[r] = m;
  }
  if ((lane & 15) == 0)
#pragma unroll
    for (int r = 0; r < 8; ++r) red[wave][r + 8 * hi] = pm[r];
  __syncthreads();
  if (threadIdx.x < 16) {
    float m = red[0][threadIdx.x];
#pragma unroll
    for (int w = 1; w < 8; ++w) m = fmaxf(m, red[w][threadIdx.x]);
    rowmax[threadIdx.x] = m;
  }
  __syncthreads();
  // row sum(exp)
#pragma unroll
  for (int r = 0; r < 8; ++r) {
    float rm = rowmax[r + 8 * hi];
    float s = 0.f;
#pragma unroll
    for (int t = 0; t < 8; ++t) s += __expf(acc[t][r] - rm);
    for (int off = 1; off < 16; off <<= 1) s += __shfl_xor(s, off, 32);
    pm[r] = s;
  }
  if ((lane & 15) == 0)
#pragma unroll
    for (int r = 0; r < 8; ++r) red[wave][r + 8 * hi] = pm[r];
  __syncthreads();
  if (threadIdx.x < 16) {
    float s = red[0][threadIdx.x];
#pragma unroll
    for (int w = 1; w < 8; ++w) s += red[w][threadIdx.x];
    lse[threadIdx.x] = rowmax[threadIdx.x] + __logf(s);
  }
  __syncthreads();
#pragma unroll
  for (int t = 0; t < 8; ++t) {
    int n = (wave * 8 + t) * 16 + (lane & 15);
#pragma unroll
    for (int r = 0; r < 8; ++r) {
      int m = r + 8 * hi;
      size_t o = (size_t)(blockIdx.x * 16 + m) * NLO + n;
      float v = acc[t][r] - lse[m];
      ulp[o] = v;
      align_out[o] = __expf(v);
    }
  }
}

// ---------------- conv GEMM: ctx[c] = sum_tap E_shift @ Wk + b  (bf16 out) ----------------
// WG = 8 waves; wave owns 4 N-tiles, WG covers CONV_MB M-tiles: each B-fragment
// feeds CONV_MB WMMAs, each A-fragment feeds 4 WMMAs.
__global__ void __launch_bounds__(256) k_conv(const unsigned short* __restrict__ e_lin,  // [NV*14, ND]
                                              const unsigned short* __restrict__ w_bf,   // [2][3][ND][ND]
                                              const float* __restrict__ conv_b,
                                              const float* __restrict__ ins_b,
                                              unsigned short* __restrict__ ctx) {        // [2][NM, ND]
  int c = blockIdx.y;
  int wave = threadIdx.x >> 5, lane = threadIdx.x & 31;
  int mA = lane & 15, hi = lane >> 4;
  size_t abase[CONV_MB];
#pragma unroll
  for (int mt = 0; mt < CONV_MB; ++mt) {
    int rg = (blockIdx.x * CONV_MB + mt) * 16 + mA;
    int v = rg / NL, l = rg - v * NL;
    abase[mt] = (size_t)(v * 14 + l) * ND;      // + tap*ND gives e[v, l+tap-1]
  }
  const unsigned short* W = w_bf + (size_t)c * 3 * ND * ND;
  v8f acc[CONV_MB][4];
  v8f z = {0.f,0.f,0.f,0.f,0.f,0.f,0.f,0.f};
#pragma unroll
  for (int mt = 0; mt < CONV_MB; ++mt)
#pragma unroll
    for (int t = 0; t < 4; ++t) acc[mt][t] = z;

  for (int kc = 0; kc < ND / 32; ++kc) {
#pragma unroll
    for (int tap = 0; tap < 3; ++tap) {
      v16bf a[CONV_MB];
#pragma unroll
      for (int mt = 0; mt < CONV_MB; ++mt) {
        const unsigned short* ap = e_lin + abase[mt] + (size_t)tap * ND + kc * 32 + hi * 16;
        a[mt] = load_a_frag(ap);
        if (kc + 1 < ND / 32) __builtin_prefetch(ap + 32, 0, 3);  // global_prefetch_b8
      }
#pragma unroll
      for (int t = 0; t < 4; ++t) {
        int nt = wave * 4 + t;
        v16bf b = load_b_frag(W + ((size_t)tap * ND + nt * 16 + mA) * ND + kc * 32 + hi * 16);
#pragma unroll
        for (int mt = 0; mt < CONV_MB; ++mt)
          acc[mt][t] = wmma_bf16(a[mt], b, acc[mt][t]);
      }
    }
  }
  const float* bias = c ? ins_b : conv_b;
  unsigned short* out = ctx + (size_t)c * NM * ND;
#pragma unroll
  for (int t = 0; t < 4; ++t) {
    int n = (wave * 4 + t) * 16 + (lane & 15);
    float bv = bias[n];
#pragma unroll
    for (int mt = 0; mt < CONV_MB; ++mt)
#pragma unroll
      for (int r = 0; r < 8; ++r) {
        size_t row = (size_t)(blockIdx.x * CONV_MB + mt) * 16 + r + 8 * hi;
        out[row * ND + n] = f32_to_bf16(acc[mt][t][r] + bv);
      }
  }
}

// ---------------- costs: logits, log_softmax, interp with gathered global lp ----------------
// WG covers COST_MB M-tiles (full LO row each, for the softmax); each loaded
// B-fragment feeds COST_MB WMMAs -> halves lue traffic from L2.
__global__ void __launch_bounds__(256) k_costs(const unsigned short* __restrict__ ctx,  // [2][NM, ND]
                                               const unsigned short* __restrict__ lue,  // [NLO, ND]
                                               const float* __restrict__ ulp,           // [NK, NLO]
                                               const int* __restrict__ seg,             // [NV, NL]
                                               float* __restrict__ out) {               // sub | ins
  __shared__ float red[8][16 * COST_MB];
  __shared__ float rowmax[16 * COST_MB];
  __shared__ float lse[16 * COST_MB];
  int c = blockIdx.y;
  int wave = threadIdx.x >> 5, lane = threadIdx.x & 31;
  int mA = lane & 15, hi = lane >> 4;
  const unsigned short* A = ctx + (size_t)c * NM * ND;
  v8f acc[COST_MB][8];
  v8f z = {0.f,0.f,0.f,0.f,0.f,0.f,0.f,0.f};
#pragma unroll
  for (int mt = 0; mt < COST_MB; ++mt)
#pragma unroll
    for (int t = 0; t < 8; ++t) acc[mt][t] = z;

  for (int kc = 0; kc < ND / 32; ++kc) {
    v16bf a[COST_MB];
#pragma unroll
    for (int mt = 0; mt < COST_MB; ++mt) {
      const unsigned short* ap =
          A + (size_t)((blockIdx.x * COST_MB + mt) * 16 + mA) * ND + kc * 32 + hi * 16;
      a[mt] = load_a_frag(ap);
      if (kc + 1 < ND / 32) __builtin_prefetch(ap + 32, 0, 3);    // global_prefetch_b8
    }
#pragma unroll
    for (int t = 0; t < 8; ++t) {
      int nt = wave * 8 + t;
      v16bf b = load_b_frag(lue + (size_t)(nt * 16 + mA) * ND + kc * 32 + hi * 16);
#pragma unroll
      for (int mt = 0; mt < COST_MB; ++mt)
        acc[mt][t] = wmma_bf16(a[mt], b, acc[mt][t]);
    }
  }
  // ---- log-softmax stats over the full LO row (8 waves x 8 N-tiles) ----
  float pm[COST_MB][8];
#pragma unroll
  for (int mt = 0; mt < COST_MB; ++mt)
#pragma unroll
    for (int r = 0; r < 8; ++r) {
      float m = acc[mt][0][r];
#pragma unroll
      for (int t = 1; t < 8; ++t) m = fmaxf(m, acc[mt][t][r]);
      for (int off = 1; off < 16; off <<= 1) m = fmaxf(m, __shfl_xor(m, off, 32));
      pm[mt][r] = m;
    }
  if ((lane & 15) == 0)
#pragma unroll
    for (int mt = 0; mt < COST_MB; ++mt)
#pragma unroll
      for (int r = 0; r < 8; ++r) red[wave][mt * 16 + r + 8 * hi] = pm[mt][r];
  __syncthreads();
  if (threadIdx.x < 16 * COST_MB) {
    float m = red[0][threadIdx.x];
#pragma unroll
    for (int w = 1; w < 8; ++w) m = fmaxf(m, red[w][threadIdx.x]);
    rowmax[threadIdx.x] = m;
  }
  __syncthreads();
#pragma unroll
  for (int mt = 0; mt < COST_MB; ++mt)
#pragma unroll
    for (int r = 0; r < 8; ++r) {
      float rm = rowmax[mt * 16 + r + 8 * hi];
      float s = 0.f;
#pragma unroll
      for (int t = 0; t < 8; ++t) s += __expf(acc[mt][t][r] - rm);
      for (int off = 1; off < 16; off <<= 1) s += __shfl_xor(s, off, 32);
      pm[mt][r] = s;
    }
  if ((lane & 15) == 0)
#pragma unroll
    for (int mt = 0; mt < COST_MB; ++mt)
#pragma unroll
      for (int r = 0; r < 8; ++r) red[wave][mt * 16 + r + 8 * hi] = pm[mt][r];
  __syncthreads();
  if (threadIdx.x < 16 * COST_MB) {
    float s = red[0][threadIdx.x];
#pragma unroll
    for (int w = 1; w < 8; ++w) s += red[w][threadIdx.x];
    lse[threadIdx.x] = rowmax[threadIdx.x] + __logf(s);
  }
  __syncthreads();
  // ---- epilogue: gather unit_log_probs rows, interpolate, negate ----
  float* dst = out + (size_t)c * NM * NLO;
#pragma unroll
  for (int mt = 0; mt < COST_MB; ++mt) {
    int srow[8];
#pragma unroll
    for (int r = 0; r < 8; ++r) {
      int rg = (blockIdx.x * COST_MB + mt) * 16 + r + 8 * hi;
      int v = rg / NL, l = rg - v * NL;
      srow[r] = seg[v * NL + l];
    }
#pragma unroll
    for (int t = 0; t < 8; ++t) {
      int n = (wave * 8 + t) * 16 + (lane & 15);
#pragma unroll
      for (int r = 0; r < 8; ++r) {
        int m = r + 8 * hi;
        size_t rg = (size_t)(blockIdx.x * COST_MB + mt) * 16 + m;
        float glp = ulp[(size_t)srow[r] * NLO + n];
        float clp = acc[mt][t][r] - lse[mt * 16 + m];
        dst[rg * NLO + n] = -(CTX_W * clp + PLAIN_W * glp);
      }
    }
  }
}

extern "C" void kernel_launch(void* const* d_in, const int* in_sizes, int n_in,
                              void* d_out, int out_size, void* d_ws, size_t ws_size,
                              hipStream_t stream) {
  const float* kue_f   = (const float*)d_in[0];   // [NK, ND]
  const float* ua_f    = (const float*)d_in[1];   // [NLO, NK]
  const float* conv_w  = (const float*)d_in[2];   // [ND, ND, 3]
  const float* conv_b  = (const float*)d_in[3];   // [ND]
  const float* insw_f  = (const float*)d_in[4];   // [ND, ND, 3]
  const float* ins_b   = (const float*)d_in[5];   // [ND]
  const int*   seg     = (const int*)d_in[6];     // [NV, NL]
  const int*   vlen    = (const int*)d_in[7];     // [NV]
  float* out = (float*)d_out;                     // sub | ins | alignment

  // workspace layout (bytes)
  char* ws = (char*)d_ws;
  size_t off = 0;
  unsigned short* kue_bf  = (unsigned short*)(ws + off); off += (size_t)NK * ND * 2;
  unsigned short* kueT_bf = (unsigned short*)(ws + off); off += (size_t)ND * NK * 2;
  unsigned short* ua_bf   = (unsigned short*)(ws + off); off += (size_t)NLO * NK * 2;
  unsigned short* w_bf    = (unsigned short*)(ws + off); off += (size_t)2 * 3 * ND * ND * 2;
  unsigned short* lue_bf  = (unsigned short*)(ws + off); off += (size_t)NLO * ND * 2;
  float*          ulp     = (float*)(ws + off);          off += (size_t)NK * NLO * 4;
  unsigned short* e_lin   = (unsigned short*)(ws + off); off += (size_t)NV * 14 * ND * 2;
  unsigned short* ctx     = (unsigned short*)(ws + off); off += (size_t)2 * NM * ND * 2;
  (void)ws_size; (void)in_sizes; (void)n_in; (void)out_size;

  // prep: f32 -> bf16 conversions / repacks
  k_f32_to_bf16<<<(NK * ND + 255) / 256, 256, 0, stream>>>(kue_f, kue_bf, NK * ND);
  k_f32_to_bf16<<<(NLO * NK + 255) / 256, 256, 0, stream>>>(ua_f, ua_bf, NLO * NK);
  k_transpose_kue<<<(ND * NK + 255) / 256, 256, 0, stream>>>(kue_f, kueT_bf);
  k_pack_w<<<(3 * ND * ND + 255) / 256, 256, 0, stream>>>(conv_w, w_bf);
  k_pack_w<<<(3 * ND * ND + 255) / 256, 256, 0, stream>>>(insw_f, w_bf + (size_t)3 * ND * ND);

  // lost_unit_emb (bf16, row-major [LO, D] == B^T for all logits GEMMs)
  k_gemm_lue<<<256, 256, 0, stream>>>(ua_bf, kueT_bf, lue_bf);

  // unit_log_probs + alignment output
  float* align_out = out + (size_t)2 * NM * NLO;
  k_unit_logprobs<<<NK / 16, 256, 0, stream>>>(kue_bf, lue_bf, ulp, align_out);

  // padded, masked, gathered embedding matrix for the conv GEMMs
  k_build_elin<<<NV * 14, 256, 0, stream>>>(kue_bf, seg, vlen, e_lin);

  // both convolutions as shifted GEMMs, bias fused, bf16 out (M-blocked x4)
  k_conv<<<dim3(NMT / CONV_MB, 2), 256, 0, stream>>>(e_lin, w_bf, conv_b, ins_b, ctx);

  // context logits + log_softmax + interpolation -> sub / ins (M-blocked x2)
  k_costs<<<dim3(NMT / COST_MB, 2), 256, 0, stream>>>(ctx, lue_bf, ulp, seg, out);
}
